// EGNNLayer_7275674599708
// MI455X (gfx1250) — matrix operand order repaired and checked by reading, not compiled, not run on hardware
//
#include <hip/hip_runtime.h>
#include <hip/hip_bf16.h>
#include <math.h>

#define BATCH 2
#define NPTS  512
#define DIM   128
#define HID   256

typedef _Float16 h8  __attribute__((ext_vector_type(8)));
typedef _Float16 h16 __attribute__((ext_vector_type(16)));
typedef float    f8  __attribute__((ext_vector_type(8)));

// Fast SiLU: v * rcp(1 + e^-v). v_rcp_f32 (1 trans op) instead of the
// ~8-op IEEE divide sequence; error ~1 ulp, far below f16-WMMA quantization.
__device__ __forceinline__ float silu(float v) {
    return v * __builtin_amdgcn_rcpf(1.0f + __expf(-v));
}

// ---------------------------------------------------------------------------
// Kernel 0: pack ew2 (256x256 f32, row-major [k][n]) into f16 B-fragment order.
// Fragment (nt, c): B tile is K = c*32..c*32+31, N = nt*16..nt*16+15.
// Contiguous-k B layout (same k order as load_a): lane l holds column n = nt*16 + (l&15),
//   K = c*32 + ((l&16)?16:0) + 2p + hb  for p=0..7 (VGPR index), hb=0/1.
// Packed linear index: (((nt*8 + c)*32 + lane)*16 + 2p + hb) -> 32B/lane, coalesced.
// ---------------------------------------------------------------------------
__global__ void pack_ew2_kernel(const float* __restrict__ ew2,
                                _Float16* __restrict__ out) {
    int idx = blockIdx.x * 256 + threadIdx.x;      // 65536 total
    int ph   = idx & 15;
    int p    = ph >> 1;
    int hb   = ph & 1;
    int lane = (idx >> 4) & 31;
    int c    = (idx >> 9) & 7;
    int nt   = idx >> 12;
    int n = nt * 16 + (lane & 15);
    int k = c * 32 + ((lane & 16) ? 16 : 0) + 2 * p + hb;
    out[idx] = (_Float16)ew2[k * HID + n];
}

// ---------------------------------------------------------------------------
// Kernel 1: HA[bn,k] = sum_d h[bn,d]*ew1[d,k]; HB uses ew1[128+d,k].
// ---------------------------------------------------------------------------
__global__ __launch_bounds__(256) void ha_hb_kernel(const float* __restrict__ h,
                                                    const float* __restrict__ ew1,
                                                    float* __restrict__ HA,
                                                    float* __restrict__ HB) {
    int bn = blockIdx.x;            // 0 .. B*N-1
    int k  = threadIdx.x;           // 0 .. 255
    __shared__ float hrow[DIM];
    if (k < DIM) hrow[k] = h[bn * DIM + k];
    __syncthreads();
    float a = 0.0f, bsum = 0.0f;
#pragma unroll 4
    for (int d = 0; d < DIM; ++d) {
        float hv = hrow[d];
        a    += hv * ew1[d * HID + k];
        bsum += hv * ew1[(DIM + d) * HID + k];
    }
    HA[bn * HID + k] = a;
    HB[bn * HID + k] = bsum;
}

// ---------------------------------------------------------------------------
// Kernel 2: edge MLP + message sum, WMMA f16 16x16x32.
// One block per (b, i). 256 threads = 8 waves. Wave w owns N-tiles {2w, 2w+1}.
// Loop over 32 j-tiles: build e1 (16 x 256) f16 in LDS, GEMM vs packed ew2
// (B fragments reloaded from global for every j-tile and K chunk), A fragments
// double-buffered from LDS so ds_load latency overlaps WMMA issue, bias+SiLU
// on the f32 accumulators, running message sum in registers.
// ---------------------------------------------------------------------------
__global__ __launch_bounds__(256) void edge_kernel(
    const float*    __restrict__ x,     // (B,N,3)
    const float*    __restrict__ HA,    // (B*N,256)
    const float*    __restrict__ HB,    // (B*N,256)
    const float*    __restrict__ ew1,   // (257,256): row 256 = dist weights
    const float*    __restrict__ eb1,   // (256)
    const _Float16* __restrict__ ew2p,  // packed B fragments
    const float*    __restrict__ eb2,   // (256)
    float*          __restrict__ msg)   // (B*N,256)
{
    const int b = blockIdx.x >> 9;
    const int i = blockIdx.x & (NPTS - 1);
    const int t = threadIdx.x;
    const int lane = t & 31;
    const int wave = t >> 5;

    __shared__ _Float16 e1[16][264];    // padded stride: conflict-light ds_load_b128
    __shared__ float sdist[16];

    // e1-producer mapping: thread handles row r (j within tile), k-group kg.
    const int r  = t & 15;
    const int kg = t >> 4;              // 0..15, covers k = kg*16 .. kg*16+15
    float hab[16], wd[16];              // hab = HA_i[k] + eb1[k] (folded)
    const float* HAi = HA + (b * NPTS + i) * HID;
#pragma unroll
    for (int q = 0; q < 16; ++q) {
        int k = kg * 16 + q;
        hab[q] = HAi[k] + eb1[k];
        wd[q]  = ew1[256 * HID + k];
    }
    const float xi0 = x[(b * NPTS + i) * 3 + 0];
    const float xi1 = x[(b * NPTS + i) * 3 + 1];
    const float xi2 = x[(b * NPTS + i) * 3 + 2];

    // WMMA-consumer setup
    const int nt0 = wave * 2;
    const float bias0 = eb2[nt0 * 16 + (lane & 15)];
    const float bias1 = eb2[(nt0 + 1) * 16 + (lane & 15)];
    f8 msum0 = {}, msum1 = {};

    const int arow = lane & 15;
    const int asel = (lane >> 4) * 16;  // lanes 16-31 take K offset +16

    // A-fragment loader (16-bit A 16x32 layout): lane arow, K
    //   [c*32+asel .. +15] contiguous
    auto load_a = [&](int c) -> h16 {
        const _Float16* ap = &e1[arow][c * 32 + asel];
        h8 alo = *(const h8*)ap;
        h8 ahi = *(const h8*)(ap + 8);
        h16 a;
#pragma unroll
        for (int q = 0; q < 8; ++q) { a[q] = alo[q]; a[8 + q] = ahi[q]; }
        return a;
    };

    for (int jt = 0; jt < NPTS / 16; ++jt) {
        __syncthreads();                 // protect e1 rewrite vs prior reads
        if (t < 16) {
            int j = jt * 16 + t;
            float d0 = x[(b * NPTS + j) * 3 + 0] - xi0;
            float d1 = x[(b * NPTS + j) * 3 + 1] - xi1;
            float d2 = x[(b * NPTS + j) * 3 + 2] - xi2;
            d0 -= rintf(d0); d1 -= rintf(d1); d2 -= rintf(d2);  // minimum image
            sdist[t] = sqrtf(d0 * d0 + d1 * d1 + d2 * d2 + 1e-8f);
        }
        __syncthreads();

        // e1 = silu(HA_i + HB_j + dist*wd + eb1), f32 math -> f16 to LDS
        {
            const float* HBr = HB + (b * NPTS + jt * 16 + r) * HID + kg * 16;
            if (jt + 1 < NPTS / 16) {
                __builtin_prefetch(HBr + 16 * HID, 0, 1);   // next j-tile rows
            }
            float dist = sdist[r];
#pragma unroll
            for (int q = 0; q < 16; ++q) {
                float v = hab[q] + HBr[q] + dist * wd[q];
                e1[r][kg * 16 + q] = (_Float16)silu(v);
            }
        }
        __syncthreads();

        // GEMM: (16 x 256) @ (256 x 32) per wave, K in 8 chunks of 32.
        // Double-buffer A so ds_load for step c+1 issues before WMMAs of step c.
        f8 acc0 = {}, acc1 = {};
        h16 a_cur = load_a(0);
#pragma unroll
        for (int c = 0; c < 8; ++c) {
            h16 a_next;
            if (c < 7) a_next = load_a(c + 1);

            const h8* bp0 = (const h8*)(ew2p + (((nt0) * 8 + c) * 32 + lane) * 16);
            h16 b0; { h8 l0 = bp0[0], l1 = bp0[1];
#pragma unroll
                      for (int q = 0; q < 8; ++q) { b0[q] = l0[q]; b0[8 + q] = l1[q]; } }
            const h8* bp1 = (const h8*)(ew2p + (((nt0 + 1) * 8 + c) * 32 + lane) * 16);
            h16 b1; { h8 l0 = bp1[0], l1 = bp1[1];
#pragma unroll
                      for (int q = 0; q < 8; ++q) { b1[q] = l0[q]; b1[8 + q] = l1[q]; } }

            acc0 = __builtin_amdgcn_wmma_f32_16x16x32_f16(
                       false, a_cur, false, b0, (short)0, acc0, false, false);
            acc1 = __builtin_amdgcn_wmma_f32_16x16x32_f16(
                       false, a_cur, false, b1, (short)0, acc1, false, false);
            a_cur = a_next;
        }

        // e2 = silu(acc + eb2); accumulate message sum (rows = j's -> summed later)
#pragma unroll
        for (int q = 0; q < 8; ++q) {
            msum0[q] += silu(acc0[q] + bias0);
            msum1[q] += silu(acc1[q] + bias1);
        }
    }

    // Reduce the 16 fragment rows: VGPRs 0..7 hold rows {0..7} (lanes 0-15)
    // or rows {8..15} (lanes 16-31), column n = nt*16 + (lane&15).
    float s0 = 0.0f, s1 = 0.0f;
#pragma unroll
    for (int q = 0; q < 8; ++q) { s0 += msum0[q]; s1 += msum1[q]; }
    s0 += __shfl_xor(s0, 16, 32);
    s1 += __shfl_xor(s1, 16, 32);
    if (lane < 16) {
        msg[(b * NPTS + i) * HID + nt0 * 16 + lane]       = s0;
        msg[(b * NPTS + i) * HID + (nt0 + 1) * 16 + lane] = s1;
    }
}

// ---------------------------------------------------------------------------
// Kernel 3: node MLP. out = h + silu([h,msg]@nw1 + nb1)@nw2 + nb2
// ---------------------------------------------------------------------------
__global__ __launch_bounds__(256) void node_kernel(const float* __restrict__ h,
                                                   const float* __restrict__ msg,
                                                   const float* __restrict__ nw1,
                                                   const float* __restrict__ nb1,
                                                   const float* __restrict__ nw2,
                                                   const float* __restrict__ nb2,
                                                   float* __restrict__ out) {
    int bn = blockIdx.x;
    int t  = threadIdx.x;
    __shared__ float ni[DIM + HID];   // 384
    __shared__ float t1[HID];
    if (t < DIM) ni[t] = h[bn * DIM + t];
    ni[DIM + t] = msg[bn * HID + t];
    __syncthreads();
    float acc = nb1[t];
#pragma unroll 4
    for (int d = 0; d < DIM + HID; ++d) acc += ni[d] * nw1[d * HID + t];
    t1[t] = silu(acc);
    __syncthreads();
    if (t < DIM) {
        float o = nb2[t];
#pragma unroll 4
        for (int j = 0; j < HID; ++j) o += t1[j] * nw2[j * DIM + t];
        out[bn * DIM + t] = h[bn * DIM + t] + o;
    }
}

// ---------------------------------------------------------------------------
// Kernel 4: second tuple output = x passthrough
// ---------------------------------------------------------------------------
__global__ void copy_x_kernel(const float* __restrict__ x, float* __restrict__ out) {
    int idx = blockIdx.x * 256 + threadIdx.x;
    if (idx < BATCH * NPTS * 3) out[BATCH * NPTS * DIM + idx] = x[idx];
}

extern "C" void kernel_launch(void* const* d_in, const int* in_sizes, int n_in,
                              void* d_out, int out_size, void* d_ws, size_t ws_size,
                              hipStream_t stream) {
    const float* h       = (const float*)d_in[0];   // (2,512,128)
    const float* x       = (const float*)d_in[1];   // (2,512,3)
    // d_in[2] = lattice (unused by the math path)
    const float* ew1     = (const float*)d_in[3];   // (257,256)
    const float* eb1     = (const float*)d_in[4];   // (256)
    const float* ew2     = (const float*)d_in[5];   // (256,256)
    const float* eb2     = (const float*)d_in[6];   // (256)
    const float* nw1     = (const float*)d_in[7];   // (384,256)
    const float* nb1     = (const float*)d_in[8];   // (256)
    const float* nw2     = (const float*)d_in[9];   // (256,128)
    const float* nb2     = (const float*)d_in[10];  // (128)
    float* out = (float*)d_out;

    float* ws = (float*)d_ws;
    float*    HA   = ws;                       // 2*512*256 f32
    float*    HB   = ws + BATCH * NPTS * HID;  // 2*512*256 f32
    float*    msg  = ws + 2 * BATCH * NPTS * HID;
    _Float16* ew2p = (_Float16*)(ws + 3 * BATCH * NPTS * HID);  // 65536 f16

    pack_ew2_kernel<<<HID * HID / 256, 256, 0, stream>>>(ew2, ew2p);
    ha_hb_kernel<<<BATCH * NPTS, 256, 0, stream>>>(h, ew1, HA, HB);
    edge_kernel<<<BATCH * NPTS, 256, 0, stream>>>(x, HA, HB, ew1, eb1, ew2p, eb2, msg);
    node_kernel<<<BATCH * NPTS, 256, 0, stream>>>(h, msg, nw1, nb1, nw2, nb2, out);
    copy_x_kernel<<<(BATCH * NPTS * 3 + 255) / 256, 256, 0, stream>>>(x, out);
}
